// DualAttention_47364899340726
// MI455X (gfx1250) — matrix operand order repaired and checked by reading, not compiled, not run on hardware
//
#include <hip/hip_runtime.h>
#include <hip/hip_bf16.h>

// ---------------------------------------------------------------------------
// DualAttention (strict-causal, two value tensors) for gfx1250 (MI455X).
//   Pre-pass: K -> bf16 (same layout), V1/V2 -> per-head transposed bf16.
//   Main:     transposed flash attention (S^T = K*Q^T, O^T = V^T*P^T) with
//             double-buffered Tensor-Data-Mover tile loads (tensor_load_to_lds
//             + s_wait_tensorcnt), 24x v_wmma_f32_16x16x32_bf16 per 64-key
//             block per wave, in-lane softmax (single xor-16 exchange),
//             raw v_exp_f32 exponentials (args always <= 0).
// B=2, S=2048, D=1024, H=16 -> dh=64.  Workspace: 24 MB bf16 staging in d_ws.
// ---------------------------------------------------------------------------

#define B_  2
#define S_  2048
#define D_  1024
#define H_  16
#define DH_ 64
#define LOG2E 1.4426950408889634f

typedef __attribute__((ext_vector_type(16))) __bf16 v16bf;
typedef __attribute__((ext_vector_type(8)))  __bf16 v8bf;
typedef __attribute__((ext_vector_type(8)))  float  v8f;
typedef __attribute__((ext_vector_type(4)))  unsigned int v4u;
typedef __attribute__((ext_vector_type(8)))  int v8i;
typedef __attribute__((ext_vector_type(4)))  int v4i;

#ifdef __has_builtin
#if __has_builtin(__builtin_amdgcn_tensor_load_to_lds)
#define USE_TDM 1
#endif
#if __has_builtin(__builtin_amdgcn_exp2f)
#define HAS_RAW_EXP2 1
#endif
#endif
#ifndef USE_TDM
#define USE_TDM 0
#endif
#ifndef HAS_RAW_EXP2
#define HAS_RAW_EXP2 0
#endif

// Raw v_exp_f32: our arguments are always <= 0 (x - rowmax), and the hugely
// negative masked scores must underflow to exactly 0 -- no libm range fixup.
__device__ __forceinline__ float fast_exp2(float x) {
#if HAS_RAW_EXP2
    return __builtin_amdgcn_exp2f(x);
#else
    return exp2f(x);
#endif
}

__device__ __forceinline__ unsigned pack_bf16(float x, float y) {
    union { unsigned u; __bf16 h[2]; } t;
    t.h[0] = (__bf16)x; t.h[1] = (__bf16)y;
    return t.u;
}

// A-operand fragment (16x32 bf16) from two contiguous 16B LDS chunks.
__device__ __forceinline__ v16bf afrag(const __bf16* p0, const __bf16* p1) {
    v8bf lo = *(const v8bf*)p0;
    v8bf hi = *(const v8bf*)p1;
    v16bf a;
#pragma unroll
    for (int e = 0; e < 8; ++e) { a[e] = lo[e]; a[8 + e] = hi[e]; }
    return a;
}

#if USE_TDM
// 2-D TDM load: tile_d1 rows x tile_d0 elements (bf16), row stride stride0
// elements, into LDS at byte offset lds_off.  D# per CDNA5 ISA.
// 6-arg builtin: (uint32x4 g0, int32x8 g1, int32x4 g2, int32x4 g3,
//                 int32x8 gx, i32 cpol)
__device__ __forceinline__ void tdm_load_2d(unsigned lds_off, const __bf16* gptr,
                                            unsigned tensor_d0, unsigned tensor_d1,
                                            unsigned tile_d0, unsigned tile_d1,
                                            unsigned long long stride0) {
    const unsigned long long ga = (unsigned long long)(uintptr_t)gptr;
    v4u g0;
    g0[0] = 1u;                                          // count=1, user D#
    g0[1] = lds_off;                                     // LDS byte address
    g0[2] = (unsigned)(ga & 0xFFFFFFFFu);                // global_addr[31:0]
    g0[3] = (unsigned)((ga >> 32) & 0x01FFFFFFu)         // global_addr[56:32]
          | (2u << 30);                                  // type=2 ("image")
    v8i g1;
    g1[0] = (int)(1u << 16);                             // data_size=1 (2B)
    g1[1] = (int)((tensor_d0 & 0xFFFFu) << 16);          // tensor_dim0[15:0]
    g1[2] = (int)((tensor_d0 >> 16) | ((tensor_d1 & 0xFFFFu) << 16));
    g1[3] = (int)((tensor_d1 >> 16) | (tile_d0 << 16));  // tile_dim0
    g1[4] = (int)(tile_d1 & 0xFFFFu);                    // tile_dim1 (dim2=0)
    g1[5] = (int)(unsigned)(stride0 & 0xFFFFFFFFu);      // dim0_stride[31:0]
    g1[6] = (int)(unsigned)((stride0 >> 32) & 0xFFFFu);  // dim0_stride[47:32]
    g1[7] = 0;
    v4i z4 = (v4i)0;                                     // <=2D: groups 2/3 null
    v8i z8 = (v8i)0;
    __builtin_amdgcn_tensor_load_to_lds(g0, g1, z4, z4, z8, 0);
}
#endif

// ---------------------------------------------------------------------------
// Pre-pass 1: elementwise K f32 -> bf16 (layout preserved).
__global__ __launch_bounds__(256)
void cvt_k_bf16(const float* __restrict__ src, __bf16* __restrict__ dst) {
    const size_t i = ((size_t)blockIdx.x * 256 + threadIdx.x) * 8;
#pragma unroll
    for (int e = 0; e < 8; e += 2)
        *(unsigned*)&dst[i + e] = pack_bf16(src[i + e], src[i + e + 1]);
}

// Pre-pass 2: V1/V2 f32 [b][s][h*64+d] -> bf16 transposed [bh][d][s].
__global__ __launch_bounds__(256)
void cvt_vT_bf16(const float* __restrict__ v1, const float* __restrict__ v2,
                 __bf16* __restrict__ t1, __bf16* __restrict__ t2) {
    const int d  = threadIdx.x & 63;
    const int sg = threadIdx.x >> 6;
    const int s0 = blockIdx.x * 32 + sg * 8;
    const int bh = blockIdx.y;
    const int b  = bh >> 4, h = bh & 15;
    const size_t src0 = ((size_t)(b * S_ + s0)) * D_ + h * DH_ + d;
    const size_t dst0 = ((size_t)bh * DH_ + d) * S_ + s0;
#pragma unroll
    for (int e = 0; e < 8; e += 2) {
        *(unsigned*)&t1[dst0 + e] =
            pack_bf16(v1[src0 + (size_t)e * D_], v1[src0 + (size_t)(e + 1) * D_]);
        *(unsigned*)&t2[dst0 + e] =
            pack_bf16(v2[src0 + (size_t)e * D_], v2[src0 + (size_t)(e + 1) * D_]);
    }
}

// ---------------------------------------------------------------------------
__global__ __launch_bounds__(256)
void DualAttention_47364899340726_kernel(const float* __restrict__ q,
                                         const __bf16* __restrict__ kb,
                                         const __bf16* __restrict__ vt1,
                                         const __bf16* __restrict__ vt2,
                                         float* __restrict__ o1,
                                         float* __restrict__ o2) {
    // Double-buffered bf16 tiles: K [key 64][feat 64], V^T [feat 64][key 64]
    __shared__ __bf16 sK  [2][64 * DH_];    // 16 KB
    __shared__ __bf16 sV1T[2][DH_ * 64];    // 16 KB
    __shared__ __bf16 sV2T[2][DH_ * 64];    // 16 KB

    const int tid  = threadIdx.x;
    const int lane = tid & 31;
    const int lgrp = lane >> 4;
    const int lcol = lane & 15;

    const int bh = blockIdx.y;
    const int b  = bh >> 4, h = bh & 15;

    const int tile_base = blockIdx.x * 128;
    const int q0        = tile_base + (tid >> 5) * 16;
    const int iq        = q0 + lcol;            // lane's query row (S^T: N=query)

    const size_t head_off = (size_t)h * DH_;
    const __bf16* kb_bh  = kb  + ((size_t)b * S_) * D_ + head_off;   // [S][D] rows
    const __bf16* vt1_bh = vt1 + (size_t)bh * DH_ * S_;              // [64][S] rows
    const __bf16* vt2_bh = vt2 + (size_t)bh * DH_ * S_;

    // ---- Q B-fragments (loop invariant, f32 global -> bf16) ---------------
    const float* qrow = q + ((size_t)(b * S_ + iq)) * D_ + head_off;
    v16bf qb0, qb1;
    {
        const float sc = 0.125f * LOG2E;        // 1/sqrt(dh), log2-domain exp
        const int fo = lgrp * 8;
#pragma unroll
        for (int e = 0; e < 16; ++e) {
            qb0[e] = (__bf16)(sc * qrow[fo + e + (e >= 8 ? 8 : 0)]);
            qb1[e] = (__bf16)(sc * qrow[32 + fo + e + (e >= 8 ? 8 : 0)]);
        }
    }

    v8f acc1[4], acc2[4];
#pragma unroll
    for (int g = 0; g < 4; ++g) { acc1[g] = (v8f)0.0f; acc2[g] = (v8f)0.0f; }
    float m = -1e30f, l = 0.0f;

    const int nblk = tile_base / 64 + 2;        // 64-key blocks, strict-causal

#if USE_TDM
    if (tid < 32) {                             // wave 0 issues the DMA batch 0
        tdm_load_2d((unsigned)(uintptr_t)&sK[0][0],   kb_bh,      64, 64, 64, 64, D_);
        tdm_load_2d((unsigned)(uintptr_t)&sV1T[0][0], vt1_bh,     64, 64, 64, 64, S_);
        tdm_load_2d((unsigned)(uintptr_t)&sV2T[0][0], vt2_bh,     64, 64, 64, 64, S_);
    }
#endif

    for (int blk = 0; blk < nblk; ++blk) {
        const int j0 = blk * 64;
        const int p  = blk & 1;

#if USE_TDM
        __syncthreads();                        // buf[1-p] free for prefetch
        if (tid < 32) {
            if (blk + 1 < nblk) {
                const int jn = (blk + 1) * 64;
                const int pn = 1 - p;
                tdm_load_2d((unsigned)(uintptr_t)&sK[pn][0],
                            kb_bh + (size_t)jn * D_, 64, 64, 64, 64, D_);
                tdm_load_2d((unsigned)(uintptr_t)&sV1T[pn][0],
                            vt1_bh + jn,             64, 64, 64, 64, S_);
                tdm_load_2d((unsigned)(uintptr_t)&sV2T[pn][0],
                            vt2_bh + jn,             64, 64, 64, 64, S_);
                __builtin_amdgcn_s_wait_tensorcnt(3);   // batch blk complete
            } else {
                __builtin_amdgcn_s_wait_tensorcnt(0);
            }
        }
        __syncthreads();                        // buf[p] visible to all waves
#else
        __syncthreads();
        {   // fallback: plain bf16 tile copy (no conversion needed)
            const int row = tid >> 2;           // 0..63
            const int cb  = (tid & 3) * 16;     // 16 bf16 = 32 B per thread
            const __bf16* ks = kb_bh  + (size_t)(j0 + row) * D_ + cb;
            const __bf16* a1 = vt1_bh + (size_t)row * S_ + j0 + cb;
            const __bf16* a2 = vt2_bh + (size_t)row * S_ + j0 + cb;
            *(v8bf*)&sK  [p][row * DH_ + cb]     = *(const v8bf*)ks;
            *(v8bf*)&sK  [p][row * DH_ + cb + 8] = *(const v8bf*)(ks + 8);
            *(v8bf*)&sV1T[p][row * 64 + cb]      = *(const v8bf*)a1;
            *(v8bf*)&sV1T[p][row * 64 + cb + 8]  = *(const v8bf*)(a1 + 8);
            *(v8bf*)&sV2T[p][row * 64 + cb]      = *(const v8bf*)a2;
            *(v8bf*)&sV2T[p][row * 64 + cb + 8]  = *(const v8bf*)(a2 + 8);
        }
        __syncthreads();
#endif

        // ---- S^T(64x16) = K(64x64) * Q^T(64x16): 8 WMMAs ------------------
        v8f st[4];
#pragma unroll
        for (int kg = 0; kg < 4; ++kg) {
            const __bf16* krow = &sK[p][(16 * kg + lcol) * DH_ + lgrp * 8];
            v16bf ka0 = afrag(krow,      krow + 16);    // feats  0..31
            v16bf ka1 = afrag(krow + 32, krow + 48);    // feats 32..63
            v8f c = (v8f)0.0f;
            c = __builtin_amdgcn_wmma_f32_16x16x32_bf16(false, ka0, false, qb0, (short)0, c, false, false);
            c = __builtin_amdgcn_wmma_f32_16x16x32_bf16(false, ka1, false, qb1, (short)0, c, false, false);
            st[kg] = c;
        }

        // ---- strict-causal mask (diagonal blocks only, wave-uniform) ------
        if (j0 + 64 > q0) {
#pragma unroll
            for (int kg = 0; kg < 4; ++kg)
#pragma unroll
                for (int r = 0; r < 8; ++r) {
                    const int j = j0 + 16 * kg + r + 8 * lgrp;
                    st[kg][r] = (j < iq) ? st[kg][r] : -1e9f;
                }
        }

        // ---- online softmax: in-lane reduction + one xor-16 exchange ------
        float bm = -1e30f;
#pragma unroll
        for (int kg = 0; kg < 4; ++kg)
#pragma unroll
            for (int r = 0; r < 8; ++r) bm = fmaxf(bm, st[kg][r]);
        bm = fmaxf(bm, __shfl_xor(bm, 16));

        const float mn   = fmaxf(m, bm);
        const float corr = fast_exp2(m - mn);
        m = mn;

        float p_[4][8];
        float ls = 0.0f;
#pragma unroll
        for (int kg = 0; kg < 4; ++kg)
#pragma unroll
            for (int r = 0; r < 8; ++r) {
                p_[kg][r] = fast_exp2(st[kg][r] - mn);
                ls += p_[kg][r];
            }
        ls += __shfl_xor(ls, 16);
        l = l * corr + ls;
#pragma unroll
        for (int g = 0; g < 4; ++g) {
            acc1[g] = acc1[g] * (v8f)corr;
            acc2[g] = acc2[g] * (v8f)corr;
        }

        // ---- P^T B-fragments packed in-lane (xor-16 results unused) -----------------
        v16bf pb[2];
#pragma unroll
        for (int c = 0; c < 2; ++c) {
            unsigned a0[4], a1[4], x0[4], x1[4];
#pragma unroll
            for (int j2 = 0; j2 < 4; ++j2) {
                a0[j2] = pack_bf16(p_[2 * c    ][2 * j2], p_[2 * c    ][2 * j2 + 1]);
                a1[j2] = pack_bf16(p_[2 * c + 1][2 * j2], p_[2 * c + 1][2 * j2 + 1]);
            }
#pragma unroll
            for (int j2 = 0; j2 < 4; ++j2) {
                x0[j2] = __shfl_xor(a0[j2], 16);
                x1[j2] = __shfl_xor(a1[j2], 16);
            }
            union { unsigned u[8]; v16bf hh; } pu;
#pragma unroll
            for (int j2 = 0; j2 < 4; ++j2) {
                pu.u[j2]     = a0[j2];  // K 0-7  / 8-15
                pu.u[4 + j2] = a1[j2];  // K 16-23 / 24-31
            }
            pb[c] = pu.hh;
        }

        // ---- O^T += V^T(64x64) * P^T(64x16): 16 WMMAs ---------------------
#pragma unroll
        for (int g = 0; g < 4; ++g) {
            const __bf16* vr1 = &sV1T[p][(16 * g + lcol) * 64 + lgrp * 8];
            const __bf16* vr2 = &sV2T[p][(16 * g + lcol) * 64 + lgrp * 8];
            v16bf va;
            va = afrag(vr1,      vr1 + 16);
            acc1[g] = __builtin_amdgcn_wmma_f32_16x16x32_bf16(false, va, false, pb[0], (short)0, acc1[g], false, false);
            va = afrag(vr1 + 32, vr1 + 48);
            acc1[g] = __builtin_amdgcn_wmma_f32_16x16x32_bf16(false, va, false, pb[1], (short)0, acc1[g], false, false);
            va = afrag(vr2,      vr2 + 16);
            acc2[g] = __builtin_amdgcn_wmma_f32_16x16x32_bf16(false, va, false, pb[0], (short)0, acc2[g], false, false);
            va = afrag(vr2 + 32, vr2 + 48);
            acc2[g] = __builtin_amdgcn_wmma_f32_16x16x32_bf16(false, va, false, pb[1], (short)0, acc2[g], false, false);
        }
    }

    // ---- epilogue: O^T layout -> contiguous dh columns per lane -----------
    const float inv = 1.0f / ((l > 0.0f) ? l : 1.0f);
    const size_t obase = ((size_t)(b * S_ + iq)) * D_ + head_off + 8 * lgrp;
#pragma unroll
    for (int g = 0; g < 4; ++g) {
        float4 w0, w1;
        w0.x = acc1[g][0] * inv; w0.y = acc1[g][1] * inv;
        w0.z = acc1[g][2] * inv; w0.w = acc1[g][3] * inv;
        w1.x = acc1[g][4] * inv; w1.y = acc1[g][5] * inv;
        w1.z = acc1[g][6] * inv; w1.w = acc1[g][7] * inv;
        *(float4*)(o1 + obase + 16 * g)     = w0;
        *(float4*)(o1 + obase + 16 * g + 4) = w1;
        w0.x = acc2[g][0] * inv; w0.y = acc2[g][1] * inv;
        w0.z = acc2[g][2] * inv; w0.w = acc2[g][3] * inv;
        w1.x = acc2[g][4] * inv; w1.y = acc2[g][5] * inv;
        w1.z = acc2[g][6] * inv; w1.w = acc2[g][7] * inv;
        *(float4*)(o2 + obase + 16 * g)     = w0;
        *(float4*)(o2 + obase + 16 * g + 4) = w1;
    }
}

extern "C" void kernel_launch(void* const* d_in, const int* in_sizes, int n_in,
                              void* d_out, int out_size, void* d_ws, size_t ws_size,
                              hipStream_t stream) {
    const float* q  = (const float*)d_in[0];
    const float* k  = (const float*)d_in[1];
    const float* v1 = (const float*)d_in[2];
    const float* v2 = (const float*)d_in[3];
    // d_in[4] = num_attn_heads (fixed 16, baked into the kernel)

    float* o1 = (float*)d_out;
    float* o2 = (float*)d_out + (size_t)B_ * S_ * D_;

    const size_t N = (size_t)B_ * S_ * D_;      // 4.19M elements
    __bf16* kb  = (__bf16*)d_ws;                // 8 MB
    __bf16* vt1 = kb  + N;                      // 8 MB
    __bf16* vt2 = vt1 + N;                      // 8 MB  (24 MB total in d_ws)

    cvt_k_bf16 <<<dim3((unsigned)(N / (256 * 8))), dim3(256), 0, stream>>>(k, kb);
    cvt_vT_bf16<<<dim3(S_ / 32, B_ * H_),          dim3(256), 0, stream>>>(v1, v2, vt1, vt2);

    dim3 grid(S_ / 128, B_ * H_);   // (16, 32)
    dim3 block(256);                // 8 wave32 waves
    DualAttention_47364899340726_kernel<<<grid, block, 0, stream>>>(q, kb, vt1, vt2, o1, o2);
}
